// LocalPoolPointnet_23192823398639
// MI455X (gfx1250) — matrix operand order repaired and checked by reading, not compiled, not run on hardware
//
#include <hip/hip_runtime.h>
#include <hip/hip_bf16.h>
#include <stdint.h>

// ---------------------------------------------------------------------------
// LocalPoolPointnet + UNet3D for MI455X (gfx1250).
//  - Point MLP / scatter ops: VALU + native u32/f32 atomics (3.2 GFLOP, trivial)
//  - All 3D convs: implicit GEMM via v_wmma_f32_16x16x32_f16 (171 GFLOP total)
// ---------------------------------------------------------------------------

#define BB 2
#define TT 30000
#define RR 64
#define R3C (RR*RR*RR)

typedef _Float16 v16h __attribute__((ext_vector_type(16)));
typedef _Float16 v8h  __attribute__((ext_vector_type(8)));
typedef float    v8f  __attribute__((ext_vector_type(8)));

// ---- order-preserving float <-> uint encoding for atomicMax scatter-max ----
__device__ __forceinline__ unsigned fenc(float f) {
  unsigned u = __float_as_uint(f);
  return (u & 0x80000000u) ? ~u : (u | 0x80000000u);
}
__device__ __forceinline__ float fdec(unsigned k) {
  unsigned u = (k & 0x80000000u) ? (k & 0x7fffffffu) : ~k;
  return __uint_as_float(u);
}
#define NEGINF_KEY 0x007FFFFFu   // fenc(-inf)

// ---------------------------------------------------------------------------
// fills
// ---------------------------------------------------------------------------
__global__ void fill_u32_kernel(unsigned* p, unsigned v, size_t n) {
  size_t i = (size_t)blockIdx.x * blockDim.x + threadIdx.x;
  if (i < n) p[i] = v;
}
__global__ void fill_f32_kernel(float* p, float v, size_t n) {
  size_t i = (size_t)blockIdx.x * blockDim.x + threadIdx.x;
  if (i < n) p[i] = v;
}

// ---------------------------------------------------------------------------
// ResnetBlockFC(2H -> H): net = relu(x)@W0+b0; dx = relu(net)@W1+b1; out = x@Ws+dx
// weights come from LDS (uniform broadcast)
// ---------------------------------------------------------------------------
__device__ void resblock_dev(const float* x, const float* W0, const float* b0,
                             const float* W1, const float* b1, const float* Ws,
                             float* o) {
  float rx[64];
#pragma unroll
  for (int k = 0; k < 64; ++k) rx[k] = fmaxf(x[k], 0.0f);
  float h[32];
  for (int j = 0; j < 32; ++j) {
    float s = b0[j];
    for (int k = 0; k < 64; ++k) s += rx[k] * W0[k*32 + j];
    h[j] = fmaxf(s, 0.0f);               // relu for the second linear
  }
  for (int j = 0; j < 32; ++j) {
    float s = b1[j];
    for (int k = 0; k < 32; ++k) s += h[k] * W1[k*32 + j];
    float sc = 0.0f;
    for (int k = 0; k < 64; ++k) sc += x[k] * Ws[k*32 + j];
    o[j] = s + sc;
  }
}

// ---------------------------------------------------------------------------
// fc_pos + block0 + voxel index
// ---------------------------------------------------------------------------
__global__ __launch_bounds__(256)
void point_init_kernel(const float* __restrict__ p,
                       const float* __restrict__ Wp, const float* __restrict__ bp,
                       const float* __restrict__ W0, const float* __restrict__ b0,
                       const float* __restrict__ W1, const float* __restrict__ b1,
                       const float* __restrict__ Ws,
                       float* __restrict__ net, int* __restrict__ idx) {
  __shared__ float sWp[192], sbp[64], sW0[2048], sb0[32], sW1[1024], sb1[32], sWs[2048];
  for (int i = threadIdx.x; i < 192;  i += 256) sWp[i] = Wp[i];
  for (int i = threadIdx.x; i < 64;   i += 256) sbp[i] = bp[i];
  for (int i = threadIdx.x; i < 2048; i += 256) sW0[i] = W0[i];
  for (int i = threadIdx.x; i < 32;   i += 256) sb0[i] = b0[i];
  for (int i = threadIdx.x; i < 1024; i += 256) sW1[i] = W1[i];
  for (int i = threadIdx.x; i < 32;   i += 256) sb1[i] = b1[i];
  for (int i = threadIdx.x; i < 2048; i += 256) sWs[i] = Ws[i];
  __syncthreads();
  int t = blockIdx.x * 256 + threadIdx.x;
  if (t >= BB * TT) return;

  float px = p[3*t+0], py = p[3*t+1], pz = p[3*t+2];
  const float inv = 1.0f / (1.0f + 0.1f + 1e-3f);
  float nx = fminf(fmaxf(px*inv + 0.5f, 0.0f), 1.0f - 1e-3f);
  float ny = fminf(fmaxf(py*inv + 0.5f, 0.0f), 1.0f - 1e-3f);
  float nz = fminf(fmaxf(pz*inv + 0.5f, 0.0f), 1.0f - 1e-3f);
  int ix = (int)floorf(nx * 64.0f);
  int iy = (int)floorf(ny * 64.0f);
  int iz = (int)floorf(nz * 64.0f);
  idx[t] = ix + 64 * (iy + 64 * iz);

  float x[64];
#pragma unroll
  for (int j = 0; j < 64; ++j)
    x[j] = sbp[j] + px*sWp[j] + py*sWp[64+j] + pz*sWp[128+j];
  float o[32];
  resblock_dev(x, sW0, sb0, sW1, sb1, sWs, o);
#pragma unroll
  for (int j = 0; j < 32; ++j) net[(size_t)t*32 + j] = o[j];
}

// ---------------------------------------------------------------------------
// scatter-max of net[BT,32] into mg[b][vox][32] (encoded u32)
// ---------------------------------------------------------------------------
__global__ void scatter_max_kernel(const float* __restrict__ net,
                                   const int* __restrict__ idx,
                                   unsigned* __restrict__ mg) {
  int e = blockIdx.x * 256 + threadIdx.x;
  if (e >= BB * TT * 32) return;
  int t = e >> 5, c = e & 31;
  int b = t / TT;
  size_t g = ((size_t)b * R3C + idx[t]) * 32 + c;
  atomicMax(&mg[g], fenc(net[e]));
}

// ---------------------------------------------------------------------------
// gather pooled + resblock (in-place on net)
// ---------------------------------------------------------------------------
__global__ __launch_bounds__(256)
void point_block_kernel(float* __restrict__ net,
                        const unsigned* __restrict__ mg, const int* __restrict__ idx,
                        const float* __restrict__ W0, const float* __restrict__ b0,
                        const float* __restrict__ W1, const float* __restrict__ b1,
                        const float* __restrict__ Ws) {
  __shared__ float sW0[2048], sb0[32], sW1[1024], sb1[32], sWs[2048];
  for (int i = threadIdx.x; i < 2048; i += 256) sW0[i] = W0[i];
  for (int i = threadIdx.x; i < 32;   i += 256) sb0[i] = b0[i];
  for (int i = threadIdx.x; i < 1024; i += 256) sW1[i] = W1[i];
  for (int i = threadIdx.x; i < 32;   i += 256) sb1[i] = b1[i];
  for (int i = threadIdx.x; i < 2048; i += 256) sWs[i] = Ws[i];
  __syncthreads();
  int t = blockIdx.x * 256 + threadIdx.x;
  if (t >= BB * TT) return;

  float x[64];
#pragma unroll
  for (int c = 0; c < 32; ++c) x[c] = net[(size_t)t*32 + c];
  const unsigned* g = &mg[((size_t)(t / TT) * R3C + idx[t]) * 32];
#pragma unroll
  for (int c = 0; c < 32; ++c) x[32 + c] = fdec(g[c]);
  float o[32];
  resblock_dev(x, sW0, sb0, sW1, sb1, sWs, o);
#pragma unroll
  for (int j = 0; j < 32; ++j) net[(size_t)t*32 + j] = o[j];
}

// ---------------------------------------------------------------------------
// fc_c + scatter-mean accumulation (sums in NCDHW grid layout + counts)
// ---------------------------------------------------------------------------
__global__ __launch_bounds__(256)
void fc_scatter_kernel(const float* __restrict__ net, const int* __restrict__ idx,
                       const float* __restrict__ Wc, const float* __restrict__ bc,
                       float* __restrict__ g0, float* __restrict__ cnt) {
  __shared__ float sW[1024], sb[32];
  for (int i = threadIdx.x; i < 1024; i += 256) sW[i] = Wc[i];
  for (int i = threadIdx.x; i < 32;   i += 256) sb[i] = bc[i];
  __syncthreads();
  int t = blockIdx.x * 256 + threadIdx.x;
  if (t >= BB * TT) return;
  int b = t / TT;
  int vox = idx[t];
  float x[32];
#pragma unroll
  for (int c = 0; c < 32; ++c) x[c] = net[(size_t)t*32 + c];
  for (int j = 0; j < 32; ++j) {
    float s = sb[j];
    for (int k = 0; k < 32; ++k) s += x[k] * sW[k*32 + j];
    atomicAdd(&g0[((size_t)(b*32 + j)) * R3C + vox], s);
  }
  atomicAdd(&cnt[(size_t)b * R3C + vox], 1.0f);
}

__global__ void normalize_kernel(float* __restrict__ g0, const float* __restrict__ cnt) {
  size_t e = (size_t)blockIdx.x * 256 + threadIdx.x;
  if (e >= (size_t)BB * R3C) return;
  int b = (int)(e / R3C);
  size_t vox = e % R3C;
  float inv = 1.0f / fmaxf(cnt[e], 1.0f);
  for (int c = 0; c < 32; ++c)
    g0[((size_t)(b*32 + c)) * R3C + vox] *= inv;
}

// ---------------------------------------------------------------------------
// 2x2x2 maxpool, NCDHW
// ---------------------------------------------------------------------------
__global__ void maxpool2_kernel(const float* __restrict__ in, float* __restrict__ out,
                                int C, int S) {
  int h = S >> 1;
  size_t total = (size_t)BB * C * h * h * h;
  size_t e = (size_t)blockIdx.x * 256 + threadIdx.x;
  if (e >= total) return;
  int x = (int)(e % h); size_t r = e / h;
  int y = (int)(r % h); r /= h;
  int z = (int)(r % h); r /= h;
  int c = (int)(r % C); int b = (int)(r / C);
  const float* ip = &in[(((size_t)(b*C + c) * S + 2*z) * S + 2*y) * S + 2*x];
  size_t SS = (size_t)S * S;
  float m = ip[0];
  m = fmaxf(m, ip[1]);
  m = fmaxf(m, ip[S]);      m = fmaxf(m, ip[S + 1]);
  m = fmaxf(m, ip[SS]);     m = fmaxf(m, ip[SS + 1]);
  m = fmaxf(m, ip[SS + S]); m = fmaxf(m, ip[SS + S + 1]);
  out[e] = m;
}

// ---------------------------------------------------------------------------
// Weight pre-pack: w[Co][Ci][taps] f32 -> per-lane f16 B fragments.
// Fragment f = (tap*nchunks + chunk)*ntN + nt, 32 lanes x 16 halves each.
// Lane L holds B[k=(L/16)*8+(j%8)+(j/8)*16][n=L%16] for j=0..15 (v_wmma 16x16x32 B layout).
// ---------------------------------------------------------------------------
__global__ void prepack_kernel(const float* __restrict__ w, _Float16* __restrict__ bp,
                               int Ci, int Co, int taps) {
  int total = taps * Ci * Co;
  int e = blockIdx.x * 256 + threadIdx.x;
  if (e >= total) return;
  int j = e & 15;
  int lane = (e >> 4) & 31;
  int f = e >> 9;
  int ntN = Co >> 4, nch = Ci >> 5;
  int nt = f % ntN;
  int rest = f / ntN;
  int chunk = rest % nch;
  int t = rest / nch;
  int n = lane & 15;
  int co = nt*16 + n;
  int k = ((lane >> 4) << 3) + (j & 7) + ((j >> 3) << 4);
  int ci = (chunk << 5) + k;
  bp[e] = (_Float16)w[((size_t)co * Ci + ci) * taps + t];
}

// ---------------------------------------------------------------------------
// Implicit-GEMM conv3d (SAME, 3x3x3 or 1x1x1) with v_wmma_f32_16x16x32_f16.
// One block per (b, z, y) output row. 8 waves, each one 16(x) x 16(co) tile.
// Input channels read from srcA (optionally 2x nearest-up) then srcB (skip).
// LDS tile layout: As[(row*Sp + xi)*32 + ci_local], f16.
// ---------------------------------------------------------------------------
template <int TAPS>
__global__ __launch_bounds__(256)
void conv3d_wmma_kernel(const float* __restrict__ srcA, int CA, int upA,
                        const float* __restrict__ srcB, int CB,
                        const _Float16* __restrict__ bpack,
                        const float* __restrict__ bias,
                        float* __restrict__ out,
                        int S, int Co, int doRelu) {
  constexpr int LDSH = (TAPS == 3) ? (9 * 66 * 32) : (64 * 32);
  __shared__ alignas(32) _Float16 As[LDSH];

  const int b = blockIdx.y;
  const int z = blockIdx.x / S;
  const int y = blockIdx.x % S;
  const int Sp   = (TAPS == 3) ? (S + 2) : S;
  const int rows = (TAPS == 3) ? 9 : 1;
  const int Ci  = CA + CB;
  const int nch = Ci >> 5;
  const int ntN = Co >> 4;
  const int ntM = S >> 4;
  const int hS  = S >> 1;

  const int tid  = threadIdx.x;
  const int wave = tid >> 5;
  const int lane = tid & 31;
  const bool active = wave < ntM * ntN;   // == 8 tiles for every shipped shape
  const int mt = wave % ntM;
  const int nt = wave / ntM;

  v8f acc = {};
  const int total = rows * Sp * 32;

  for (int chunk = 0; chunk < nch; ++chunk) {
    __syncthreads();
    for (int e = tid; e < total; e += 256) {
      int c   = e & 31;
      int xi  = (e >> 5) % Sp;
      int row = (e >> 5) / Sp;
      int zz, yy, xx;
      if (TAPS == 3) { zz = z + row/3 - 1; yy = y + row%3 - 1; xx = xi - 1; }
      else           { zz = z;             yy = y;             xx = xi;     }
      float v = 0.0f;
      if (zz >= 0 && zz < S && yy >= 0 && yy < S && xx >= 0 && xx < S) {
        int ci = (chunk << 5) + c;
        if (ci < CA) {
          v = upA ? srcA[(((size_t)(b*CA + ci) * hS + (zz >> 1)) * hS + (yy >> 1)) * hS + (xx >> 1)]
                  : srcA[(((size_t)(b*CA + ci) * S  +  zz      ) * S  +  yy      ) * S  +  xx      ];
        } else {
          v = srcB[(((size_t)(b*CB + (ci - CA)) * S + zz) * S + yy) * S + xx];
        }
      }
      As[e] = (_Float16)v;
    }
    __syncthreads();

    if (active) {                         // uniform per wave -> EXEC all ones
      const int m  = lane & 15;
      const int kb = (lane >> 4) << 3;    // 0 or 8: A-layout K base
      const int x0 = mt*16 + m;
      const int ntaps = (TAPS == 3) ? 27 : 1;
      for (int t = 0; t < ntaps; ++t) {
        const int row = (TAPS == 3) ? (t / 3) : 0;   // kz*3+ky
        const int xk  = (TAPS == 3) ? (t % 3) : 0;   // kx
        const _Float16* ap = &As[(row * Sp + x0 + xk) * 32 + kb];
        v8h alo = *(const v8h*)(ap);
        v8h ahi = *(const v8h*)(ap + 16);
        v16h a = __builtin_shufflevector(alo, ahi,
                                         0,1,2,3,4,5,6,7,8,9,10,11,12,13,14,15);
        const _Float16* bptr =
            bpack + (size_t)((t * nch + chunk) * ntN + nt) * 512 + lane * 16;
        v16h bf = *(const v16h*)bptr;
        acc = __builtin_amdgcn_wmma_f32_16x16x32_f16(false, a, false, bf,
                                                     (short)0, acc, false, false);
      }
    }
  }

  if (active) {
    const int n  = lane & 15;
    const int co = nt*16 + n;
    const float bv = bias[co];
    const int xbase = mt*16 + ((lane >> 4) << 3);  // C/D layout: m = r + 8*(lane/16)
    float* op = &out[(((size_t)(b*Co + co) * S + z) * S + y) * S + xbase];
    v8f res;
#pragma unroll
    for (int r = 0; r < 8; ++r) {
      float v = acc[r] + bv;
      res[r] = doRelu ? fmaxf(v, 0.0f) : v;
    }
    *(v8f*)op = res;
  }
}

// ---------------------------------------------------------------------------
// host orchestration
// ---------------------------------------------------------------------------
extern "C" void kernel_launch(void* const* d_in, const int* in_sizes, int n_in,
                              void* d_out, int out_size, void* d_ws, size_t ws_size,
                              hipStream_t stream) {
  (void)n_in; (void)out_size; (void)ws_size;
  auto in = [&](int i) { return (const float*)d_in[i]; };

  // JAX pytrees sort dict keys; an insertion-order harness keeps fc_pos.W (192)
  // at slot 1, a sorted harness keeps blocks[0].W0 (2048) there.
  const bool sorted = (in_sizes[1] == 2048);

  const float *P = in(0);
  const float *fcposW, *fcposb, *fccW, *fccb;
  const float *W0[5], *b0[5], *W1[5], *b1[5], *Ws[5];
  const float *we1, *be1, *we2, *be2, *we3, *be3, *wd2, *bd2, *wd1, *bd1, *wo, *bo;

  if (sorted) {
    for (int k = 0; k < 5; ++k) {
      W0[k] = in(1 + 5*k); W1[k] = in(2 + 5*k); Ws[k] = in(3 + 5*k);
      b0[k] = in(4 + 5*k); b1[k] = in(5 + 5*k);
    }
    fccW = in(26); fccb = in(27); fcposW = in(28); fcposb = in(29);
    bd1 = in(30); bd2 = in(31); be1 = in(32); be2 = in(33); be3 = in(34); bo = in(35);
    wd1 = in(36); wd2 = in(37); we1 = in(38); we2 = in(39); we3 = in(40); wo = in(41);
  } else {
    fcposW = in(1); fcposb = in(2);
    for (int k = 0; k < 5; ++k) {
      W0[k] = in(3 + 5*k); b0[k] = in(4 + 5*k); W1[k] = in(5 + 5*k);
      b1[k] = in(6 + 5*k); Ws[k] = in(7 + 5*k);
    }
    fccW = in(28); fccb = in(29);
    we1 = in(30); be1 = in(31); we2 = in(32); be2 = in(33); we3 = in(34); be3 = in(35);
    wd2 = in(36); bd2 = in(37); wd1 = in(38); bd1 = in(39); wo = in(40); bo = in(41);
  }

  // ---- workspace carve (256B aligned). "big" is time-multiplexed:
  //      maxgrid (u32) -> grid sums/means g0 -> d1 output.
  char* base = (char*)d_ws;
  size_t off = 0;
  auto carve = [&](size_t bytes) -> void* {
    void* p = base + off;
    off += (bytes + 255) & ~(size_t)255;
    return p;
  };
  int*      idx  = (int*)carve((size_t)BB*TT*4);
  float*    net  = (float*)carve((size_t)BB*TT*32*4);
  void*     big  = carve((size_t)BB*32*R3C*4);
  float*    cnt  = (float*)carve((size_t)BB*R3C*4);
  float*    e1   = (float*)carve((size_t)BB*32*R3C*4);
  float*    p1   = (float*)carve((size_t)BB*32*(R3C/8)*4);
  float*    e2f  = (float*)carve((size_t)BB*64*(R3C/8)*4);
  float*    p2   = (float*)carve((size_t)BB*64*(R3C/64)*4);
  float*    e3f  = (float*)carve((size_t)BB*128*(R3C/64)*4);
  float*    d2f  = (float*)carve((size_t)BB*64*(R3C/8)*4);
  _Float16* bp_e1 = (_Float16*)carve((size_t)27*32*32*2);
  _Float16* bp_e2 = (_Float16*)carve((size_t)27*32*64*2);
  _Float16* bp_e3 = (_Float16*)carve((size_t)27*64*128*2);
  _Float16* bp_d2 = (_Float16*)carve((size_t)27*192*64*2);
  _Float16* bp_d1 = (_Float16*)carve((size_t)27*96*32*2);
  _Float16* bp_wo = (_Float16*)carve((size_t)32*32*2);

  const int BT = BB * TT;
  dim3 blk(256);
  auto nb = [](size_t n) { return (unsigned)((n + 255) / 256); };

  // ---- weight pre-pack into WMMA B-fragment layout (f32 -> f16)
  prepack_kernel<<<nb(27*32*32),  blk, 0, stream>>>(we1, bp_e1, 32, 32, 27);
  prepack_kernel<<<nb(27*32*64),  blk, 0, stream>>>(we2, bp_e2, 32, 64, 27);
  prepack_kernel<<<nb(27*64*128), blk, 0, stream>>>(we3, bp_e3, 64, 128, 27);
  prepack_kernel<<<nb((size_t)27*192*64), blk, 0, stream>>>(wd2, bp_d2, 192, 64, 27);
  prepack_kernel<<<nb(27*96*32),  blk, 0, stream>>>(wd1, bp_d1, 96, 32, 27);
  prepack_kernel<<<nb(32*32),     blk, 0, stream>>>(wo,  bp_wo, 32, 32, 1);

  // ---- point pipeline
  point_init_kernel<<<nb(BT), blk, 0, stream>>>(P, fcposW, fcposb,
      W0[0], b0[0], W1[0], b1[0], Ws[0], net, idx);

  unsigned* mg = (unsigned*)big;
  for (int k = 1; k < 5; ++k) {
    fill_u32_kernel<<<nb((size_t)BB*R3C*32), blk, 0, stream>>>(mg, NEGINF_KEY, (size_t)BB*R3C*32);
    scatter_max_kernel<<<nb((size_t)BT*32), blk, 0, stream>>>(net, idx, mg);
    point_block_kernel<<<nb(BT), blk, 0, stream>>>(net, mg, idx,
        W0[k], b0[k], W1[k], b1[k], Ws[k]);
  }

  float* g0 = (float*)big;
  fill_f32_kernel<<<nb((size_t)BB*32*R3C), blk, 0, stream>>>(g0, 0.0f, (size_t)BB*32*R3C);
  fill_f32_kernel<<<nb((size_t)BB*R3C), blk, 0, stream>>>(cnt, 0.0f, (size_t)BB*R3C);
  fc_scatter_kernel<<<nb(BT), blk, 0, stream>>>(net, idx, fccW, fccb, g0, cnt);
  normalize_kernel<<<nb((size_t)BB*R3C), blk, 0, stream>>>(g0, cnt);

  // ---- UNet3D (all convs via WMMA implicit GEMM)
  conv3d_wmma_kernel<3><<<dim3(64*64, BB), blk, 0, stream>>>(g0, 32, 0, nullptr, 0,
      bp_e1, be1, e1, 64, 32, 1);
  maxpool2_kernel<<<nb((size_t)BB*32*32*32*32), blk, 0, stream>>>(e1, p1, 32, 64);
  conv3d_wmma_kernel<3><<<dim3(32*32, BB), blk, 0, stream>>>(p1, 32, 0, nullptr, 0,
      bp_e2, be2, e2f, 32, 64, 1);
  maxpool2_kernel<<<nb((size_t)BB*64*16*16*16), blk, 0, stream>>>(e2f, p2, 64, 32);
  conv3d_wmma_kernel<3><<<dim3(16*16, BB), blk, 0, stream>>>(p2, 64, 0, nullptr, 0,
      bp_e3, be3, e3f, 16, 128, 1);
  // d2 = conv(concat(up2(e3), e2)): channels [0,128) from up2(e3), [128,192) from e2
  conv3d_wmma_kernel<3><<<dim3(32*32, BB), blk, 0, stream>>>(e3f, 128, 1, e2f, 64,
      bp_d2, bd2, d2f, 32, 64, 1);
  // d1 = conv(concat(up2(d2), e1)) -> reuse "big" (g0 is dead now)
  conv3d_wmma_kernel<3><<<dim3(64*64, BB), blk, 0, stream>>>(d2f, 64, 1, e1, 32,
      bp_d1, bd1, (float*)big, 64, 32, 1);
  // final 1x1x1 conv, no relu, straight to output
  conv3d_wmma_kernel<1><<<dim3(64*64, BB), blk, 0, stream>>>((float*)big, 32, 0, nullptr, 0,
      bp_wo, bo, (float*)d_out, 64, 32, 0);
}
